// myExpertsBlock_30657476559561
// MI455X (gfx1250) — compile-verified, matrix operand order checked by reading, not run on hardware
//
#include <hip/hip_runtime.h>

#define HDIM 1024
#define EXPERTS 8
#define TOKENS 16384             // 4 * 4096
#define TOPK 2
#define MROWS 64                 // tokens per FFN block (4 x 16 M-subtiles)
#define CAP (2 * TOKENS + EXPERTS * MROWS)  // grouped entries incl. 64-alignment padding
#define NTILES (CAP / MROWS)     // row tiles
#define LDSP 1032                // padded LDS row pitch in bf16 (516 dwords = 4 mod 64 banks)

typedef __attribute__((ext_vector_type(4)))  unsigned int u32x4;
typedef __attribute__((ext_vector_type(2)))  unsigned int u32x2;
typedef __attribute__((ext_vector_type(4)))  float        f32x4;
typedef __attribute__((ext_vector_type(8)))  float        v8f;
typedef __attribute__((ext_vector_type(16))) __bf16       v16bf;

union Frag { v16bf v; u32x4 q[2]; };

__device__ __forceinline__ unsigned short f32_to_bf16(float f) {
  unsigned int u = __float_as_uint(f);
  u += 0x7fffu + ((u >> 16) & 1u);   // round-to-nearest-even
  return (unsigned short)(u >> 16);
}

// ---------------------------------------------------------------- init
__global__ void moe_init_kernel(float* __restrict__ y, int* __restrict__ counts,
                                int* __restrict__ perm) {
  int i = blockIdx.x * blockDim.x + threadIdx.x;   // covers TOKENS*HDIM/4
  f32x4 z = {0.f, 0.f, 0.f, 0.f};
  ((f32x4*)y)[i] = z;
  if (i < CAP) perm[i] = -1;
  if (i < EXPERTS) counts[i] = 0;
}

// ---------------------------------------------------------------- x -> bf16
__global__ void moe_cast_x_kernel(const float* __restrict__ x,
                                  unsigned short* __restrict__ xb) {
  int i = blockIdx.x * blockDim.x + threadIdx.x;   // TOKENS*HDIM/4 threads
  f32x4 v = ((const f32x4*)x)[i];
  unsigned int lo = (unsigned)f32_to_bf16(v.x) | ((unsigned)f32_to_bf16(v.y) << 16);
  unsigned int hi = (unsigned)f32_to_bf16(v.z) | ((unsigned)f32_to_bf16(v.w) << 16);
  u32x2 o = {lo, hi};
  ((u32x2*)xb)[i] = o;
}

// ---------------------------------------------------------------- w -> bf16, WMMA B-fragment order
// layout: [e][nt(64)][kt(32)][lane(32)][16 halfs], lane L holds col n = nt*16+(L&15),
// halfs 0..7 -> k = kt*32 + (L>>4)*8 + j ; halfs 8..15 -> k + 16
__global__ void moe_cast_w_kernel(const float* __restrict__ w,
                                  unsigned short* __restrict__ wb) {
  int g = blockIdx.x * blockDim.x + threadIdx.x;   // EXPERTS*64*32*32 = 524288
  int lane = g & 31;
  int kt   = (g >> 5) & 31;
  int nt   = (g >> 10) & 63;
  int e    = g >> 16;
  const float* we = w + (size_t)e * HDIM * HDIM;
  int n  = nt * 16 + (lane & 15);
  int kb = kt * 32 + ((lane >> 4) * 8);
  unsigned short tmp[16];
#pragma unroll
  for (int j = 0; j < 8; ++j) tmp[j]     = f32_to_bf16(we[(size_t)(kb + j) * HDIM + n]);
#pragma unroll
  for (int j = 0; j < 8; ++j) tmp[8 + j] = f32_to_bf16(we[(size_t)(kb + 16 + j) * HDIM + n]);
  u32x4 q0, q1;
#pragma unroll
  for (int j = 0; j < 4; ++j) q0[j] = (unsigned)tmp[2*j]   | ((unsigned)tmp[2*j+1] << 16);
#pragma unroll
  for (int j = 0; j < 4; ++j) q1[j] = (unsigned)tmp[8+2*j] | ((unsigned)tmp[9+2*j] << 16);
  u32x4* dst = (u32x4*)(wb + (size_t)g * 16);
  dst[0] = q0; dst[1] = q1;
}

// ---------------------------------------------------------------- router (one wave32 per token)
__global__ void moe_router_kernel(const float* __restrict__ x, const float* __restrict__ wr,
                                  const float* __restrict__ br, int* __restrict__ topi,
                                  float* __restrict__ topv, int* __restrict__ counts) {
  __shared__ float wr_s[HDIM * EXPERTS];
  for (int i = threadIdx.x; i < HDIM * EXPERTS; i += 256) wr_s[i] = wr[i];
  __syncthreads();
  int wave = threadIdx.x >> 5, lane = threadIdx.x & 31;
  int t = blockIdx.x * 8 + wave;
  const float* xr = x + (size_t)t * HDIM;
  float acc[EXPERTS];
#pragma unroll
  for (int e = 0; e < EXPERTS; ++e) acc[e] = 0.f;
  for (int i = 0; i < HDIM / 32; ++i) {
    int h = i * 32 + lane;
    float xv = xr[h];
#pragma unroll
    for (int e = 0; e < EXPERTS; ++e) acc[e] += xv * wr_s[h * EXPERTS + e];
  }
#pragma unroll
  for (int off = 16; off > 0; off >>= 1) {
#pragma unroll
    for (int e = 0; e < EXPERTS; ++e) acc[e] += __shfl_xor(acc[e], off, 32);
  }
  if (lane == 0) {
    float mx = -1e30f;
#pragma unroll
    for (int e = 0; e < EXPERTS; ++e) { acc[e] += br[e]; mx = fmaxf(mx, acc[e]); }
    float p[EXPERTS]; float sum = 0.f;
#pragma unroll
    for (int e = 0; e < EXPERTS; ++e) { p[e] = __expf(acc[e] - mx); sum += p[e]; }
    float inv = 1.f / sum;
    int i0 = 0; float v0 = -1.f;
#pragma unroll
    for (int e = 0; e < EXPERTS; ++e) { p[e] *= inv; if (p[e] > v0) { v0 = p[e]; i0 = e; } }
    int i1 = 0; float v1 = -1.f;
#pragma unroll
    for (int e = 0; e < EXPERTS; ++e) if (e != i0 && p[e] > v1) { v1 = p[e]; i1 = e; }
    topi[t * 2] = i0; topi[t * 2 + 1] = i1;
    topv[t * 2] = v0; topv[t * 2 + 1] = v1;
    atomicAdd(&counts[i0], 1);
    atomicAdd(&counts[i1], 1);
  }
}

// ---------------------------------------------------------------- 64-aligned expert offsets
__global__ void moe_scan_kernel(const int* __restrict__ counts, int* __restrict__ ao,
                                int* __restrict__ cursor) {
  if (threadIdx.x == 0 && blockIdx.x == 0) {
    int off = 0;
    for (int e = 0; e < EXPERTS; ++e) {
      ao[e] = off; cursor[e] = off;
      off += (counts[e] + (MROWS - 1)) & ~(MROWS - 1);
    }
    ao[EXPERTS] = off;
  }
}

// ---------------------------------------------------------------- scatter tokens into expert groups
__global__ void moe_scatter_kernel(const int* __restrict__ topi, const float* __restrict__ topv,
                                   int* __restrict__ cursor, int* __restrict__ perm,
                                   float* __restrict__ pval) {
  int t = blockIdx.x * blockDim.x + threadIdx.x;
  if (t >= TOKENS) return;
#pragma unroll
  for (int s = 0; s < TOPK; ++s) {
    int e = topi[t * 2 + s];
    int pos = atomicAdd(&cursor[e], 1);
    perm[pos] = t;
    pval[pos] = topv[t * 2 + s];
  }
}

// ---------------------------------------------------------------- grouped expert FFN, bf16 WMMA
// 512 threads = 16 wave32; 64 tokens/block; wave w owns n-tiles w*4..w*4+3 across 4 M-subtiles.
// A fragments double-buffered (a0/a1) so ds_loads never WAR the in-flight WMMA sources.
__global__ void __launch_bounds__(512, 1) moe_ffn_kernel(
    const unsigned short* __restrict__ xb, const unsigned short* __restrict__ w1b,
    const unsigned short* __restrict__ w2b, const float* __restrict__ b1,
    const float* __restrict__ b2, const int* __restrict__ perm,
    const float* __restrict__ pval, const int* __restrict__ ao, float* __restrict__ y) {
  extern __shared__ unsigned short smem[];
  unsigned short* x_s = smem;                    // 64 x LDSP bf16 (~129 KB)
  unsigned short* h_s = smem + MROWS * LDSP;     // 64 x LDSP bf16 (~129 KB)
  __shared__ int   tok_s[MROWS];
  __shared__ float wgt_s[MROWS];

  int row0 = blockIdx.x * MROWS;
  if (row0 >= ao[EXPERTS]) return;               // uniform exit: no live barriers skipped
  int e = 0;
#pragma unroll
  for (int i = 1; i < EXPERTS; ++i) if (row0 >= ao[i]) e = i;

  if (threadIdx.x < MROWS) {
    int tk = perm[row0 + threadIdx.x];
    tok_s[threadIdx.x] = tk;
    wgt_s[threadIdx.x] = (tk >= 0) ? pval[row0 + threadIdx.x] : 0.f;
  }
  __syncthreads();

  { // stage 64 rows of x: 8 threads per row, 256 B per thread, coalesced b128
    int r = threadIdx.x >> 3;
    int c = (threadIdx.x & 7) * 128;             // ushort offset within row
    int tk = tok_s[r];
    u32x4* dst = (u32x4*)&x_s[r * LDSP + c];
    if (tk >= 0) {
      const u32x4* src = (const u32x4*)&xb[(size_t)tk * HDIM + c];
#pragma unroll
      for (int j = 0; j < 16; ++j) dst[j] = src[j];
    } else {
      u32x4 z = {0, 0, 0, 0};
#pragma unroll
      for (int j = 0; j < 16; ++j) dst[j] = z;
    }
  }
  __syncthreads();

  int lane  = threadIdx.x & 31;
  int wave  = threadIdx.x >> 5;                  // 0..15
  int mrow  = lane & 15;
  int kbase = (lane >> 4) * 8;                   // A-frag K sub-base; also C-frag row base
  const v8f vzero = {0.f, 0.f, 0.f, 0.f, 0.f, 0.f, 0.f, 0.f};
  v8f acc[4][4];                                 // [m-subtile][n-tile] : 128 VGPRs

  auto load_a = [&](const unsigned short* s, int kt, Frag* a) {
#pragma unroll
    for (int mt = 0; mt < 4; ++mt) {
      int rr = mt * 16 + mrow;
      a[mt].q[0] = *(const u32x4*)&s[rr * LDSP + kt * 32 + kbase];
      a[mt].q[1] = *(const u32x4*)&s[rr * LDSP + kt * 32 + kbase + 16];
    }
  };
  auto do_mmas = [&](const unsigned short* we, int kt, const Frag* a) {
    __builtin_prefetch(we + ((size_t)(wave * 4 * 32 + kt + 2) * 32 + lane) * 16, 0, 3);
#pragma unroll
    for (int jn = 0; jn < 4; ++jn) {
      int nt = wave * 4 + jn;
      const u32x4* bp = (const u32x4*)(we + ((size_t)(nt * 32 + kt) * 32 + lane) * 16);
      Frag b; b.q[0] = bp[0]; b.q[1] = bp[1];
#pragma unroll
      for (int mt = 0; mt < 4; ++mt)                     // reuse B across 4 M-subtiles
        acc[mt][jn] = __builtin_amdgcn_wmma_f32_16x16x32_bf16(
            false, a[mt].v, false, b.v, (short)0, acc[mt][jn], false, false);
    }
  };

  const unsigned short* w1e = w1b + (size_t)e * (64 * 32 * 32 * 16);
  const unsigned short* w2e = w2b + (size_t)e * (64 * 32 * 32 * 16);

  // ---- GEMM1: h = silu(X * W1 + b1)
#pragma unroll
  for (int mt = 0; mt < 4; ++mt)
#pragma unroll
    for (int jn = 0; jn < 4; ++jn) acc[mt][jn] = vzero;
  {
    Frag a0[4], a1[4];
    load_a(x_s, 0, a0);
    for (int kt = 0; kt < 32; kt += 2) {
      load_a(x_s, kt + 1, a1);                   // prefetch next A into alternate regs
      do_mmas(w1e, kt, a0);
      if (kt + 2 < 32) load_a(x_s, kt + 2, a0);
      do_mmas(w1e, kt + 1, a1);
    }
  }
#pragma unroll
  for (int jn = 0; jn < 4; ++jn) {
    int col = (wave * 4 + jn) * 16 + mrow;
    float bias = b1[e * HDIM + col];
#pragma unroll
    for (int mt = 0; mt < 4; ++mt)
#pragma unroll
      for (int r = 0; r < 8; ++r) {
        float v = acc[mt][jn][r] + bias;
        v = v / (1.0f + __expf(-v));             // SiLU
        h_s[(mt * 16 + kbase + r) * LDSP + col] = f32_to_bf16(v);
      }
  }
  __syncthreads();

  // ---- GEMM2: o = h * W2 + b2 ; y[token] += weight * o
#pragma unroll
  for (int mt = 0; mt < 4; ++mt)
#pragma unroll
    for (int jn = 0; jn < 4; ++jn) acc[mt][jn] = vzero;
  {
    Frag a0[4], a1[4];
    load_a(h_s, 0, a0);
    for (int kt = 0; kt < 32; kt += 2) {
      load_a(h_s, kt + 1, a1);
      do_mmas(w2e, kt, a0);
      if (kt + 2 < 32) load_a(h_s, kt + 2, a0);
      do_mmas(w2e, kt + 1, a1);
    }
  }
#pragma unroll
  for (int jn = 0; jn < 4; ++jn) {
    int col = (wave * 4 + jn) * 16 + mrow;
    float bias = b2[e * HDIM + col];
#pragma unroll
    for (int mt = 0; mt < 4; ++mt)
#pragma unroll
      for (int r = 0; r < 8; ++r) {
        int m = mt * 16 + kbase + r;
        int tk = tok_s[m];
        if (tk >= 0) {
          float v = (acc[mt][jn][r] + bias) * wgt_s[m];
          unsafeAtomicAdd(&y[(size_t)tk * HDIM + col], v);  // two commutative adds per element
        }
      }
  }
}

// ---------------------------------------------------------------- host launcher
extern "C" void kernel_launch(void* const* d_in, const int* in_sizes, int n_in,
                              void* d_out, int out_size, void* d_ws, size_t ws_size,
                              hipStream_t stream) {
  (void)in_sizes; (void)n_in; (void)out_size; (void)ws_size;
  const float* x  = (const float*)d_in[0];
  const float* wr = (const float*)d_in[1];
  const float* br = (const float*)d_in[2];
  const float* w1 = (const float*)d_in[3];
  const float* b1 = (const float*)d_in[4];
  const float* w2 = (const float*)d_in[5];
  const float* b2 = (const float*)d_in[6];
  float* y = (float*)d_out;

  size_t off = 0;
  auto carve = [&](size_t bytes) {
    off = (off + 255) & ~(size_t)255;
    void* p = (char*)d_ws + off;
    off += bytes;
    return p;
  };
  unsigned short* xb  = (unsigned short*)carve((size_t)TOKENS * HDIM * 2);
  unsigned short* w1b = (unsigned short*)carve((size_t)EXPERTS * HDIM * HDIM * 2);
  unsigned short* w2b = (unsigned short*)carve((size_t)EXPERTS * HDIM * HDIM * 2);
  int*   topi   = (int*)carve((size_t)TOKENS * 2 * sizeof(int));
  float* topv   = (float*)carve((size_t)TOKENS * 2 * sizeof(float));
  int*   counts = (int*)carve(EXPERTS * sizeof(int));
  int*   ao     = (int*)carve((EXPERTS + 1) * sizeof(int));
  int*   cursor = (int*)carve(EXPERTS * sizeof(int));
  int*   perm   = (int*)carve((size_t)CAP * sizeof(int));
  float* pval   = (float*)carve((size_t)CAP * sizeof(float));

  moe_init_kernel  <<<(TOKENS * HDIM / 4) / 256, 256, 0, stream>>>(y, counts, perm);
  moe_cast_x_kernel<<<(TOKENS * HDIM / 4) / 256, 256, 0, stream>>>(x, xb);
  moe_cast_w_kernel<<<(EXPERTS * 64 * 32 * 32) / 256, 256, 0, stream>>>(w1, w1b);
  moe_cast_w_kernel<<<(EXPERTS * 64 * 32 * 32) / 256, 256, 0, stream>>>(w2, w2b);
  moe_router_kernel<<<TOKENS / 8, 256, 0, stream>>>(x, wr, br, topi, topv, counts);
  moe_scan_kernel  <<<1, 32, 0, stream>>>(counts, ao, cursor);
  moe_scatter_kernel<<<TOKENS / 256, 256, 0, stream>>>(topi, topv, cursor, perm, pval);

  size_t smem_bytes = (size_t)2 * MROWS * LDSP * sizeof(unsigned short); // ~258 KB of 320 KB WGP LDS
  moe_ffn_kernel   <<<NTILES, 512, smem_bytes, stream>>>(xb, w1b, w2b, b1, b2, perm, pval, ao, y);
}
